// GINEncoder_58583353917552
// MI455X (gfx1250) — matrix operand order repaired and checked by reading, not compiled, not run on hardware
//
#include <hip/hip_runtime.h>
#include <hip/hip_bf16.h>

typedef __bf16 bf16_t;
typedef __attribute__((ext_vector_type(16))) __bf16 v16bf;
typedef __attribute__((ext_vector_type(8)))  __bf16 v8bf;
typedef __attribute__((ext_vector_type(8)))  float   v8f;

#define HD 256   // hidden dim
#define NG 64    // num graphs

static inline int ceil_div(long long a, long long b) { return (int)((a + b - 1) / b); }

// ---------------- utility kernels ----------------

__global__ void zero_f32(float* __restrict__ p, long long n) {
  long long i = (long long)blockIdx.x * blockDim.x + threadIdx.x;
  if (i < n) p[i] = 0.0f;
}

// w: [K x Nout] f32 row-major  ->  wt: [Nout x K] bf16 row-major (transposed)
__global__ void transcvt_kernel(const float* __restrict__ w, bf16_t* __restrict__ wt,
                                int K, int Nout) {
  int i = blockIdx.x * blockDim.x + threadIdx.x;
  if (i >= K * Nout) return;
  int k = i / Nout;
  int n = i - k * Nout;
  wt[(size_t)n * K + k] = (bf16_t)w[i];
}

// agg[dst] += h[src] over edges; 4 floats per thread
__global__ void scatter_add_kernel(const float* __restrict__ h,
                                   const long long* __restrict__ ei,
                                   float* __restrict__ agg,
                                   int E, int d, int vshift) {
  long long idx = (long long)blockIdx.x * blockDim.x + threadIdx.x;
  long long total = (long long)E << vshift;
  if (idx >= total) return;
  int e  = (int)(idx >> vshift);
  int c4 = ((int)idx & ((1 << vshift) - 1)) << 2;
  int src = (int)ei[e];
  int dst = (int)ei[(size_t)E + e];
  const float4 v = *(const float4*)(h + (size_t)src * d + c4);
  float* o = agg + (size_t)dst * d + c4;
  atomicAdd(o + 0, v.x);
  atomicAdd(o + 1, v.y);
  atomicAdd(o + 2, v.z);
  atomicAdd(o + 3, v.w);
}

// zb = bf16(agg + h), elementwise
__global__ void add_cvt_kernel(const float* __restrict__ agg, const float* __restrict__ h,
                               bf16_t* __restrict__ zb, long long n) {
  long long i = (long long)blockIdx.x * blockDim.x + threadIdx.x;
  if (i < n) zb[i] = (bf16_t)(agg[i] + h[i]);
}

// ---------------- WMMA GEMM ----------------
// C[M x 256] = A[M x K] * W[K x 256] + bias ; A,BT in bf16, BT = W^T [256 x K].
// One wave computes one 16x16 tile with v_wmma_f32_16x16x32_bf16.
template<int KDIM, bool RELU, bool OUT_BF16>
__global__ void gemm_wmma_kernel(const bf16_t* __restrict__ A,
                                 const bf16_t* __restrict__ BT,
                                 const float*  __restrict__ bias,
                                 float* __restrict__ Cf,
                                 bf16_t* __restrict__ Cb,
                                 int M) {
  const int lane = threadIdx.x & 31;
  const int wave = threadIdx.x >> 5;            // 0..7
  const int half = lane >> 4;                   // 0 or 1
  const int lr   = lane & 15;
  const int m0 = blockIdx.x * 16;
  const int n0 = (blockIdx.y * 8 + wave) * 16;  // output column tile
  if (m0 >= M) return;

  const bf16_t* __restrict__ arow = A  + (size_t)(m0 + lr) * KDIM;
  const bf16_t* __restrict__ brow = BT + (size_t)(n0 + lr) * KDIM;

  v8f c = {};
#pragma unroll
  for (int kb = 0; kb < KDIM; kb += 32) {
    // 16-bit A 16x32 fragment: lanes 0-15 hold K kb+0..15 of row lr,
    //                          lanes 16-31 hold K kb+16..31
    union { v16bf v; v8bf h2[2]; } aU;
    aU.h2[0] = *(const v8bf*)(arow + kb + half * 16);
    aU.h2[1] = *(const v8bf*)(arow + kb + half * 16 + 8);
    // 16-bit B 32x16 fragment: column n0+lr, lanes 0-15 K=kb..kb+15, lanes 16-31 K=kb+16..kb+31
    v16bf b = *(const v16bf*)(brow + kb + half * 16);
    c = __builtin_amdgcn_wmma_f32_16x16x32_bf16(
        /*neg_a=*/false, aU.v, /*neg_b=*/false, b,
        /*c_mod=*/(short)0, c, /*reuse_a=*/false, /*reuse_b=*/false);
  }

  const int ocol = n0 + lr;
  const float bb = bias[ocol];
#pragma unroll
  for (int r = 0; r < 8; ++r) {
    int orow = m0 + r + half * 8;   // C/D layout: VGPR r -> M=r (lanes 0-15) / M=r+8 (lanes 16-31)
    float v = c[r] + bb;
    if (RELU) v = v > 0.0f ? v : 0.0f;
    if (OUT_BF16) Cb[(size_t)orow * HD + ocol] = (bf16_t)v;
    else          Cf[(size_t)orow * HD + ocol] = v;
  }
}

// ---------------- BatchNorm ----------------

__global__ void bn_stats_kernel(const float* __restrict__ z,
                                float* __restrict__ bnsum, float* __restrict__ bnsq,
                                int Mrows) {
  int col = threadIdx.x;               // 256 threads = 256 columns
  int r0 = blockIdx.x * 64;
  int rend = r0 + 64; if (rend > Mrows) rend = Mrows;
  float s = 0.0f, q = 0.0f;
  for (int r = r0; r < rend; ++r) {
    float v = z[(size_t)r * HD + col];
    s += v; q += v * v;
  }
  atomicAdd(&bnsum[col], s);
  atomicAdd(&bnsq[col],  q);
}

__global__ void bn_apply_relu_kernel(const float* __restrict__ z,
                                     const float* __restrict__ bnsum,
                                     const float* __restrict__ bnsq,
                                     const float* __restrict__ g,
                                     const float* __restrict__ be,
                                     float* __restrict__ h, int Mrows) {
  long long idx = (long long)blockIdx.x * blockDim.x + threadIdx.x;
  if (idx >= (long long)Mrows * HD) return;
  int col = (int)(idx & (HD - 1));
  float invN = 1.0f / (float)Mrows;
  float mu  = bnsum[col] * invN;
  float var = bnsq[col] * invN - mu * mu;
  float sc  = g[col] * rsqrtf(var + 1e-5f);
  float v = (z[idx] - mu) * sc + be[col];
  h[idx] = v > 0.0f ? v : 0.0f;
}

// ---------------- pooling ----------------

__global__ void pool_count_kernel(const long long* __restrict__ batch,
                                  float* __restrict__ counts, int Nn) {
  int i = blockIdx.x * blockDim.x + threadIdx.x;
  if (i < Nn) atomicAdd(&counts[(int)batch[i]], 1.0f);
}

__global__ void pool_sum_kernel(const float* __restrict__ h,
                                const long long* __restrict__ batch,
                                float* __restrict__ pooled, int Nn) {
  long long idx = (long long)blockIdx.x * blockDim.x + threadIdx.x;
  if (idx >= (long long)Nn * HD) return;
  int n = (int)(idx >> 8);
  int c = (int)(idx & (HD - 1));
  atomicAdd(&pooled[(size_t)batch[n] * HD + c], h[idx]);
}

__global__ void pool_div_kernel(const float* __restrict__ pooled,
                                const float* __restrict__ counts,
                                float* __restrict__ out) {
  int idx = blockIdx.x * blockDim.x + threadIdx.x;
  if (idx >= NG * HD) return;
  float cnt = counts[idx >> 8];
  out[idx] = pooled[idx] / (cnt > 1.0f ? cnt : 1.0f);
}

// ---------------- host orchestration ----------------

extern "C" void kernel_launch(void* const* d_in, const int* in_sizes, int n_in,
                              void* d_out, int out_size, void* d_ws, size_t ws_size,
                              hipStream_t stream) {
  const float*      x     = (const float*)d_in[0];
  const long long*  ei    = (const long long*)d_in[1];
  const long long*  batch = (const long long*)d_in[2];
  const float* w1_0 = (const float*)d_in[3];
  const float* b1_0 = (const float*)d_in[4];
  const float* w2_0 = (const float*)d_in[5];
  const float* b2_0 = (const float*)d_in[6];
  const float* g_0  = (const float*)d_in[7];
  const float* be_0 = (const float*)d_in[8];
  const float* w1_r = (const float*)d_in[9];
  const float* b1_r = (const float*)d_in[10];
  const float* w2_r = (const float*)d_in[11];
  const float* b2_r = (const float*)d_in[12];
  const float* g_r  = (const float*)d_in[13];
  const float* be_r = (const float*)d_in[14];

  const int Nn = in_sizes[2];             // 50000 nodes
  const int F  = in_sizes[0] / Nn;        // 128
  const int E  = in_sizes[1] / 2;         // 800000 edges
  const int H  = HD;                      // 256

  // workspace carve-up (256B aligned)
  size_t off = 0;
  char* base = (char*)d_ws;
  auto carve = [&](size_t bytes) -> char* {
    char* p = base + off;
    off += (bytes + 255) & ~(size_t)255;
    return p;
  };
  float*  h    = (float*)carve((size_t)Nn * H * sizeof(float));
  float*  agg  = (float*)carve((size_t)Nn * H * sizeof(float));  // also reused as z (GEMM2 out)
  bf16_t* zb   = (bf16_t*)carve((size_t)Nn * H * sizeof(bf16_t));
  bf16_t* y1b  = (bf16_t*)carve((size_t)Nn * H * sizeof(bf16_t));
  bf16_t* wt   = (bf16_t*)carve((size_t)(F * H + 7 * H * H) * sizeof(bf16_t));
  float*  bnsum  = (float*)carve(2 * H * sizeof(float));         // bnsum | bnsq contiguous
  float*  bnsq   = bnsum + H;
  float*  pooled = (float*)carve((size_t)(NG * H + NG) * sizeof(float)); // pooled | counts
  float*  counts = pooled + NG * H;

  bf16_t* wt0_1 = wt;                     // [H x F]
  bf16_t* wt0_2 = wt0_1 + (size_t)F * H;  // [H x H]
  bf16_t* wtr   = wt0_2 + (size_t)H * H;  // 3x {w1,w2} each [H x H]

  const int TB = 256;
  float* z = agg;  // GEMM2 output aliases agg (agg dead after add_cvt)

  // ---- weight prep (transpose + f32->bf16) ----
  transcvt_kernel<<<ceil_div((long long)F * H, TB), TB, 0, stream>>>(w1_0, wt0_1, F, H);
  transcvt_kernel<<<ceil_div((long long)H * H, TB), TB, 0, stream>>>(w2_0, wt0_2, H, H);
  for (int l = 0; l < 3; ++l) {
    transcvt_kernel<<<ceil_div((long long)H * H, TB), TB, 0, stream>>>(
        w1_r + (size_t)l * H * H, wtr + (size_t)(2 * l) * H * H, H, H);
    transcvt_kernel<<<ceil_div((long long)H * H, TB), TB, 0, stream>>>(
        w2_r + (size_t)l * H * H, wtr + (size_t)(2 * l + 1) * H * H, H, H);
  }

  dim3 gemm_grid(ceil_div(Nn, 16), (H / 16) / 8);   // (3125, 2), 8 waves/block

  // ---- layer 0 (input dim F=128) ----
  {
    zero_f32<<<ceil_div((long long)Nn * F, TB), TB, 0, stream>>>(agg, (long long)Nn * F);
    scatter_add_kernel<<<ceil_div(((long long)E * F) / 4, TB), TB, 0, stream>>>(
        x, ei, agg, E, F, 5 /* log2(128/4) */);
    add_cvt_kernel<<<ceil_div((long long)Nn * F, TB), TB, 0, stream>>>(
        agg, x, zb, (long long)Nn * F);
    gemm_wmma_kernel<128, true,  true ><<<gemm_grid, TB, 0, stream>>>(
        zb, wt0_1, b1_0, nullptr, y1b, Nn);
    gemm_wmma_kernel<256, false, false><<<gemm_grid, TB, 0, stream>>>(
        y1b, wt0_2, b2_0, z, nullptr, Nn);
    zero_f32<<<ceil_div(2 * H, TB), TB, 0, stream>>>(bnsum, 2 * H);
    bn_stats_kernel<<<ceil_div(Nn, 64), H, 0, stream>>>(z, bnsum, bnsq, Nn);
    bn_apply_relu_kernel<<<ceil_div((long long)Nn * H, TB), TB, 0, stream>>>(
        z, bnsum, bnsq, g_0, be_0, h, Nn);
  }

  // ---- layers 1..3 (H=256) ----
  for (int l = 0; l < 3; ++l) {
    const bf16_t* wA = wtr + (size_t)(2 * l) * H * H;
    const bf16_t* wB = wtr + (size_t)(2 * l + 1) * H * H;
    const float* b1 = b1_r + (size_t)l * H;
    const float* b2 = b2_r + (size_t)l * H;
    const float* gg = g_r  + (size_t)l * H;
    const float* bb = be_r + (size_t)l * H;

    zero_f32<<<ceil_div((long long)Nn * H, TB), TB, 0, stream>>>(agg, (long long)Nn * H);
    scatter_add_kernel<<<ceil_div(((long long)E * H) / 4, TB), TB, 0, stream>>>(
        h, ei, agg, E, H, 6 /* log2(256/4) */);
    add_cvt_kernel<<<ceil_div((long long)Nn * H, TB), TB, 0, stream>>>(
        agg, h, zb, (long long)Nn * H);
    gemm_wmma_kernel<256, true,  true ><<<gemm_grid, TB, 0, stream>>>(
        zb, wA, b1, nullptr, y1b, Nn);
    gemm_wmma_kernel<256, false, false><<<gemm_grid, TB, 0, stream>>>(
        y1b, wB, b2, z, nullptr, Nn);
    zero_f32<<<ceil_div(2 * H, TB), TB, 0, stream>>>(bnsum, 2 * H);
    bn_stats_kernel<<<ceil_div(Nn, 64), H, 0, stream>>>(z, bnsum, bnsq, Nn);
    bn_apply_relu_kernel<<<ceil_div((long long)Nn * H, TB), TB, 0, stream>>>(
        z, bnsum, bnsq, gg, bb, h, Nn);
  }

  // ---- global mean pool ----
  zero_f32<<<ceil_div(NG * H + NG, TB), TB, 0, stream>>>(pooled, NG * H + NG);
  pool_count_kernel<<<ceil_div(Nn, TB), TB, 0, stream>>>(batch, counts, Nn);
  pool_sum_kernel<<<ceil_div((long long)Nn * H, TB), TB, 0, stream>>>(h, batch, pooled, Nn);
  pool_div_kernel<<<ceil_div(NG * H, TB), TB, 0, stream>>>(pooled, counts, (float*)d_out);
}
